// MultiHeadAttention_15384572854551
// MI455X (gfx1250) — matrix operand order repaired and checked by reading, not compiled, not run on hardware
//
#include <hip/hip_runtime.h>

// ---------------------------------------------------------------------------
// MI455X (gfx1250) multi-head attention: all matmuls via v_wmma_f32_16x16x32_f16
// B=2, S=2048, D=1024, H=16, HD=64
// ---------------------------------------------------------------------------

#define B_SZ   2
#define S_LEN  2048
#define D_DIM  1024
#define NHEAD  16
#define HD_DIM 64
#define STOT   (B_SZ * S_LEN)   // 4096 rows total

typedef _Float16 v16h __attribute__((ext_vector_type(16)));
typedef _Float16 v8h  __attribute__((ext_vector_type(8)));
typedef float    v8f  __attribute__((ext_vector_type(8)));

// A-matrix 16x32 f16 fragment, contiguous k order (B frags use the same):
// lane<16 : VGPR0-7 = K0..15 (of row M=lane)
// lane>=16: VGPR0-7 = K16..31 (of row M=lane-16)
__device__ __forceinline__ v16h load_a_frag(const _Float16* p, int lh) {
  const v8h lo = *(const v8h*)(p + lh * 16);
  const v8h hi = *(const v8h*)(p + lh * 16 + 8);
  v16h r;
#pragma unroll
  for (int i = 0; i < 8; ++i) { r[i] = lo[i]; r[i + 8] = hi[i]; }
  return r;
}

__device__ __forceinline__ v8f wmma_f16(v16h a, v16h b, v8f c) {
  // (neg_a, A, neg_b, B, c_mod, C, reuse_a, reuse_b)
  return __builtin_amdgcn_wmma_f32_16x16x32_f16(false, a, false, b, (short)0, c,
                                                false, false);
}

// ---------------------------------------------------------------------------
// f32 -> f16 conversion (grid-stride)
// ---------------------------------------------------------------------------
__global__ __launch_bounds__(256) void f32_to_f16_kernel(
    const float* __restrict__ in, _Float16* __restrict__ out, int n) {
  int i = blockIdx.x * blockDim.x + threadIdx.x;
  int stride = gridDim.x * blockDim.x;
  for (; i < n; i += stride) out[i] = (_Float16)in[i];
}

// ---------------------------------------------------------------------------
// GEMM: C[M,N] = A[M,K] * W[N,K]^T + bias, f16 inputs, f32 accumulate.
// Block = 128 threads (4 waves). Each wave: 16 rows x 128 cols (8 N-frags).
// mode 0: f16 row-major out (scaled)   [Q/K projection]
// mode 2: f16 transposed out out[n*ldT + m]   [V projection -> Vt]
// mode 3: f32 row-major out            [final projection]
// ---------------------------------------------------------------------------
__global__ __launch_bounds__(128) void gemm_wmma_kernel(
    const _Float16* __restrict__ A, const _Float16* __restrict__ W,
    const float* __restrict__ bias, _Float16* __restrict__ outH,
    float* __restrict__ outF, int M, int N, int K, int mode, float scale,
    int ldT) {
  const int lane = threadIdx.x & 31;
  const int wave = threadIdx.x >> 5;
  const int lh = lane >> 4;   // half-wave select
  const int ln = lane & 15;   // lane-in-half = matrix row (A) / col (B,C)

  const int row0 = blockIdx.x * 64 + wave * 16;
  const int col0 = blockIdx.y * 128;

  const _Float16* aRow = A + (size_t)(row0 + ln) * K;

  v8f acc[8] = {};

  for (int k = 0; k < K; k += 32) {
    const v16h af = load_a_frag(aRow + k, lh);
#pragma unroll
    for (int j = 0; j < 8; ++j) {
      const _Float16* bp = W + (size_t)(col0 + j * 16 + ln) * K + k + lh * 16;
      const v16h bf = *(const v16h*)bp;
      acc[j] = wmma_f16(af, bf, acc[j]);
    }
  }

#pragma unroll
  for (int j = 0; j < 8; ++j) {
    const int col = col0 + j * 16 + ln;
    const float bc = bias[col];
    if (mode == 2) {
      v8h t;
#pragma unroll
      for (int v = 0; v < 8; ++v) t[v] = (_Float16)((acc[j][v] + bc) * scale);
      *(v8h*)(outH + (size_t)col * ldT + row0 + lh * 8) = t;
    } else if (mode == 0) {
#pragma unroll
      for (int v = 0; v < 8; ++v)
        outH[(size_t)(row0 + lh * 8 + v) * N + col] =
            (_Float16)((acc[j][v] + bc) * scale);
    } else {
#pragma unroll
      for (int v = 0; v < 8; ++v)
        outF[(size_t)(row0 + lh * 8 + v) * N + col] = (acc[j][v] + bc) * scale;
    }
  }
}

// ---------------------------------------------------------------------------
// Flash attention. grid = (S/64, B*H), block = 128 (4 waves).
// Each wave: 16 query rows, streams keys in chunks of 32.
// Q is pre-scaled by 1/sqrt(HD). Vt is [D][STOT] (transposed V).
// ---------------------------------------------------------------------------
__global__ __launch_bounds__(128) void attn_kernel(
    const _Float16* __restrict__ Q, const _Float16* __restrict__ Kh,
    const _Float16* __restrict__ Vt, const int* __restrict__ mask,
    _Float16* __restrict__ O) {
  __shared__ __align__(32) _Float16 pT[4][16 * 32];

  const int lane = threadIdx.x & 31;
  const int wave = threadIdx.x >> 5;
  const int lh = lane >> 4;
  const int ln = lane & 15;

  const int b = blockIdx.y >> 4;
  const int h = blockIdx.y & 15;
  const int q0 = blockIdx.x * 64 + wave * 16;

  // Q fragments for this wave's 16 rows, kept in registers for entire loop.
  const _Float16* qp = Q + (size_t)(b * S_LEN + q0 + ln) * D_DIM + h * HD_DIM;
  const v16h qa0 = load_a_frag(qp, lh);
  const v16h qa1 = load_a_frag(qp + 32, lh);

  v8f o[4] = {};
  float mr[8], lr[8];
#pragma unroll
  for (int v = 0; v < 8; ++v) { mr[v] = -1.0e30f; lr[v] = 0.0f; }

  _Float16* pw = pT[wave];

  for (int kb = 0; kb < S_LEN; kb += 32) {
    // ---- scores: S = Q(16x64) . K^T -> two 16x16 f32 fragments --------------
    const _Float16* kp0 =
        Kh + (size_t)(b * S_LEN + kb + ln) * D_DIM + h * HD_DIM + lh * 16;
    const _Float16* kp1 = kp0 + (size_t)16 * D_DIM;
    v8f s0 = {}, s1 = {};
    s0 = wmma_f16(qa0, *(const v16h*)kp0, s0);
    s0 = wmma_f16(qa1, *(const v16h*)(kp0 + 32), s0);
    s1 = wmma_f16(qa0, *(const v16h*)kp1, s1);
    s1 = wmma_f16(qa1, *(const v16h*)(kp1 + 32), s1);

    // ---- mask + online softmax ---------------------------------------------
    const int mv0 = mask[b * S_LEN + kb + ln];
    const int mv1 = mask[b * S_LEN + kb + 16 + ln];
    float mx[8];
#pragma unroll
    for (int v = 0; v < 8; ++v) {
      s0[v] = mv0 ? s0[v] : -1.0e9f;
      s1[v] = mv1 ? s1[v] : -1.0e9f;
      mx[v] = fmaxf(s0[v], s1[v]);
    }
#pragma unroll
    for (int d = 1; d < 16; d <<= 1) {
#pragma unroll
      for (int v = 0; v < 8; ++v) mx[v] = fmaxf(mx[v], __shfl_xor(mx[v], d, 32));
    }
    float al[8], ps[8];
#pragma unroll
    for (int v = 0; v < 8; ++v) {
      const float mn = fmaxf(mr[v], mx[v]);
      al[v] = __expf(mr[v] - mn);
      mr[v] = mn;
      s0[v] = __expf(s0[v] - mn);
      s1[v] = __expf(s1[v] - mn);
      ps[v] = s0[v] + s1[v];
    }
#pragma unroll
    for (int d = 1; d < 16; d <<= 1) {
#pragma unroll
      for (int v = 0; v < 8; ++v) ps[v] += __shfl_xor(ps[v], d, 32);
    }
#pragma unroll
    for (int v = 0; v < 8; ++v) lr[v] = lr[v] * al[v] + ps[v];
#pragma unroll
    for (int j = 0; j < 4; ++j)
#pragma unroll
      for (int v = 0; v < 8; ++v) o[j][v] *= al[v];

    // ---- C-layout -> A-layout transpose of P through per-wave LDS tile -----
#pragma unroll
    for (int v = 0; v < 8; ++v) {
      const int r = (v + lh * 8) * 32;
      pw[r + ln] = (_Float16)s0[v];
      pw[r + 16 + ln] = (_Float16)s1[v];
    }
    asm volatile("s_wait_dscnt 0" ::: "memory");  // LDS in-order per wave
    const v16h pa = load_a_frag(pw + ln * 32, lh);

    // ---- O += P(16x32) . V(32x64), Vt gives contiguous B-fragments ---------
    const size_t vBase = (size_t)(b * S_LEN + kb) + lh * 16;
#pragma unroll
    for (int j = 0; j < 4; ++j) {
      const _Float16* vp = Vt + (size_t)(h * HD_DIM + j * 16 + ln) * STOT + vBase;
      o[j] = wmma_f16(pa, *(const v16h*)vp, o[j]);
    }
  }

  // ---- finalize: divide by row sums, store f16 row-major -------------------
#pragma unroll
  for (int v = 0; v < 8; ++v) lr[v] = 1.0f / lr[v];
#pragma unroll
  for (int j = 0; j < 4; ++j)
#pragma unroll
    for (int v = 0; v < 8; ++v)
      O[(size_t)(b * S_LEN + q0 + lh * 8 + v) * D_DIM + h * HD_DIM + j * 16 +
        ln] = (_Float16)(o[j][v] * lr[v]);
}

// ---------------------------------------------------------------------------
// Host-side orchestration
// ---------------------------------------------------------------------------
extern "C" void kernel_launch(void* const* d_in, const int* in_sizes, int n_in,
                              void* d_out, int out_size, void* d_ws,
                              size_t ws_size, hipStream_t stream) {
  (void)in_sizes; (void)n_in; (void)out_size; (void)ws_size;

  const float* x  = (const float*)d_in[0];
  const int* mask = (const int*)d_in[1];
  const float* Wq = (const float*)d_in[2];
  const float* bq = (const float*)d_in[3];
  const float* Wk = (const float*)d_in[4];
  const float* bk = (const float*)d_in[5];
  const float* Wv = (const float*)d_in[6];
  const float* bv = (const float*)d_in[7];
  const float* Wo = (const float*)d_in[8];
  const float* bo = (const float*)d_in[9];
  float* out = (float*)d_out;

  const int BSD = B_SZ * S_LEN * D_DIM;  // 4,194,304
  const int DD = D_DIM * D_DIM;          // 1,048,576

  // Workspace layout (f16 buffers), total 48 MB.
  char* ws = (char*)d_ws;
  _Float16* xh  = (_Float16*)(ws);
  _Float16* Wqh = (_Float16*)(ws + 8388608);
  _Float16* Wkh = (_Float16*)(ws + 10485760);
  _Float16* Wvh = (_Float16*)(ws + 12582912);
  _Float16* Woh = (_Float16*)(ws + 14680064);
  _Float16* Qh  = (_Float16*)(ws + 16777216);
  _Float16* Kh  = (_Float16*)(ws + 25165824);
  _Float16* Vt  = (_Float16*)(ws + 33554432);
  _Float16* Oh  = (_Float16*)(ws + 41943040);

  f32_to_f16_kernel<<<1024, 256, 0, stream>>>(x, xh, BSD);
  f32_to_f16_kernel<<<512, 256, 0, stream>>>(Wq, Wqh, DD);
  f32_to_f16_kernel<<<512, 256, 0, stream>>>(Wk, Wkh, DD);
  f32_to_f16_kernel<<<512, 256, 0, stream>>>(Wv, Wvh, DD);
  f32_to_f16_kernel<<<512, 256, 0, stream>>>(Wo, Woh, DD);

  const dim3 gg(STOT / 64, D_DIM / 128);  // (64, 8)
  // Q = (x Wq^T + bq) * (1/sqrt(HD)), f16 row-major
  gemm_wmma_kernel<<<gg, 128, 0, stream>>>(xh, Wqh, bq, Qh, nullptr, STOT,
                                           D_DIM, D_DIM, 0, 0.125f, 0);
  // K = x Wk^T + bk, f16 row-major
  gemm_wmma_kernel<<<gg, 128, 0, stream>>>(xh, Wkh, bk, Kh, nullptr, STOT,
                                           D_DIM, D_DIM, 0, 1.0f, 0);
  // Vt = (x Wv^T + bv)^T, f16, ld = STOT
  gemm_wmma_kernel<<<gg, 128, 0, stream>>>(xh, Wvh, bv, Vt, nullptr, STOT,
                                           D_DIM, D_DIM, 2, 1.0f, STOT);

  attn_kernel<<<dim3(S_LEN / 64, B_SZ * NHEAD), 128, 0, stream>>>(Qh, Kh, Vt,
                                                                  mask, Oh);

  // out = Oh Wo^T + bo, f32
  gemm_wmma_kernel<<<gg, 128, 0, stream>>>(Oh, Woh, bo, nullptr, out, STOT,
                                           D_DIM, D_DIM, 3, 1.0f, 0);
}
